// RotaryAttention_21887153341018
// MI455X (gfx1250) — matrix operand order repaired and checked by reading, not compiled, not run on hardware
//
#include <hip/hip_runtime.h>
#include <hip/hip_bf16.h>

typedef __attribute__((ext_vector_type(16))) __bf16 v16bf;
typedef __attribute__((ext_vector_type(8)))  __bf16 v8bf;
typedef __attribute__((ext_vector_type(8)))  float  v8f;
typedef __attribute__((ext_vector_type(4)))  int    v4i;

union AFrag { v16bf v; v8bf h[2]; };

#define WMMA_BF16(a, b, c) \
  __builtin_amdgcn_wmma_f32_16x16x32_bf16(false, (a), false, (b), (short)0, (c), false, false)

#if defined(__has_builtin)
# if __has_builtin(__builtin_amdgcn_global_load_async_to_lds_b128) && \
     __has_builtin(__builtin_amdgcn_s_wait_asynccnt)
#  define HAVE_ASYNC_LDS 1
# endif
#endif
#ifndef HAVE_ASYNC_LDS
# define HAVE_ASYNC_LDS 0
#endif

// 16-byte global -> LDS copy (async on gfx1250 when available)
__device__ __forceinline__ void copy16B_g2l(const __bf16* g, __bf16* l) {
#if HAVE_ASYNC_LDS
  __builtin_amdgcn_global_load_async_to_lds_b128(
      (__attribute__((address_space(1))) v4i*)const_cast<__bf16*>(g),
      (__attribute__((address_space(3))) v4i*)l, 0, 0);
#else
  *(uint4*)l = *(const uint4*)g;
#endif
}

__device__ __forceinline__ void async_wait_all() {
#if HAVE_ASYNC_LDS
  __builtin_amdgcn_s_wait_asynccnt(0);
#endif
}

__device__ __forceinline__ v8f zero8() {
  v8f z;
#pragma unroll
  for (int i = 0; i < 8; ++i) z[i] = 0.0f;
  return z;
}

// ---------------------------------------------------------------------------
// Elementwise fp32 -> bf16 cast
// ---------------------------------------------------------------------------
__global__ void cast_f32_bf16(const float* __restrict__ in, __bf16* __restrict__ out, int n) {
  int i = blockIdx.x * blockDim.x + threadIdx.x;
  if (i < n) out[i] = (__bf16)in[i];
}

// ---------------------------------------------------------------------------
// Tiled bf16 WMMA GEMM: C[M,N] = A[M,K] * B[K,N] + bias[N]  (fp32 out)
// 256 threads = 8 waves, tile 128x128, wave = 32(M) x 64(N).
// Double-buffered LDS; A tile staged via async global->LDS copies.
// ---------------------------------------------------------------------------
__global__ __launch_bounds__(256) void gemm_bf16_wmma(
    const __bf16* __restrict__ A, const __bf16* __restrict__ B,
    const float* __restrict__ bias, float* __restrict__ C,
    int M, int N, int K) {
  __shared__ __align__(16) __bf16 As[2][128 * 40];   // [row][k] padded
  __shared__ __align__(16) __bf16 Bt[2][128 * 40];   // [n][k]   padded (transposed)

  const int tid  = threadIdx.x;
  const int lane = tid & 31;
  const int wave = tid >> 5;
  const int wm   = wave & 3;    // 0..3 -> M sub-block of 32
  const int wn   = wave >> 2;   // 0..1 -> N sub-block of 64
  const int rowBase = blockIdx.y * 128;
  const int colBase = blockIdx.x * 128;

  const int lm    = lane & 15;
  const int half  = lane >> 4;
  const int koffA = half * 8;
  const int koffB = half * 8;

  // Staging coordinates
  const int ar  = tid >> 1;          // A: row 0..127
  const int ac  = (tid & 1) * 16;    // A: col 0 or 16
  const int bkr = tid >> 3;          // B: k 0..31
  const int bnb = (tid & 7) * 16;    // B: n base 0..112

  const __bf16* aSrcBase = A + (size_t)(rowBase + ar) * K + ac;
  const __bf16* bSrcBase = B + (size_t)bkr * N + colBase + bnb;

  v8f acc[2][4];
#pragma unroll
  for (int i = 0; i < 2; ++i)
#pragma unroll
    for (int j = 0; j < 4; ++j) acc[i][j] = zero8();

  __bf16 btmp[16];

  auto stageA = [&](int k0, int buf) {
    const __bf16* src = aSrcBase + k0;
    __bf16* dst = &As[buf][ar * 40 + ac];
    copy16B_g2l(src, dst);
    copy16B_g2l(src + 8, dst + 8);
  };
  auto loadB = [&](int k0) {
    const uint4* s4 = (const uint4*)(bSrcBase + (size_t)k0 * N);
    *(uint4*)(btmp + 0) = s4[0];
    *(uint4*)(btmp + 8) = s4[1];
  };
  auto storeB = [&](int buf) {
#pragma unroll
    for (int i = 0; i < 16; ++i) Bt[buf][(bnb + i) * 40 + bkr] = btmp[i];
  };

  // Prologue: fill buffer 0
  stageA(0, 0);
  loadB(0);
  storeB(0);
  async_wait_all();
  __syncthreads();

  int cur = 0;
  for (int k0 = 0; k0 < K; k0 += 32) {
    const int nxt = cur ^ 1;
    const bool more = (k0 + 32) < K;
    if (more) {
      stageA(k0 + 32, nxt);   // async: overlaps with WMMAs below
      loadB(k0 + 32);
    }

    AFrag a[2];
#pragma unroll
    for (int i = 0; i < 2; ++i) {
      int row = wm * 32 + i * 16 + lm;
      a[i].h[0] = *(const v8bf*)(&As[cur][row * 40 + koffA]);
      a[i].h[1] = *(const v8bf*)(&As[cur][row * 40 + koffA + 16]);
    }
#pragma unroll
    for (int j = 0; j < 4; ++j) {
      AFrag bfr;
      int col = wn * 64 + j * 16 + lm;
      bfr.h[0] = *(const v8bf*)(&Bt[cur][col * 40 + koffB]);
      bfr.h[1] = *(const v8bf*)(&Bt[cur][col * 40 + koffB + 16]);
#pragma unroll
      for (int i = 0; i < 2; ++i) acc[i][j] = WMMA_BF16(a[i].v, bfr.v, acc[i][j]);
    }

    if (more) storeB(nxt);
    async_wait_all();
    __syncthreads();
    cur = nxt;
  }

  // Epilogue: C/D layout row = r + half*8, col = lm
#pragma unroll
  for (int i = 0; i < 2; ++i) {
    int mBase = rowBase + wm * 32 + i * 16;
#pragma unroll
    for (int j = 0; j < 4; ++j) {
      int nIdx = colBase + wn * 64 + j * 16 + lm;
      float bv = bias[nIdx];
#pragma unroll
      for (int r = 0; r < 8; ++r) {
        int m = mBase + r + half * 8;
        C[(size_t)m * N + nIdx] = acc[i][j][r] + bv;
      }
    }
  }
}

// ---------------------------------------------------------------------------
// RoPE for Q: [b,s,H,hd] fp32 -> [b,H,s,hd] bf16, scaled by scale^2 = 1/128
// (reference applies scale to q AND again to logits)
// ---------------------------------------------------------------------------
__global__ void rope_q_kernel(const float* __restrict__ q, __bf16* __restrict__ qb) {
  int idx  = blockIdx.x * blockDim.x + threadIdx.x;  // < 2^22
  int j    = idx & 63;
  int h    = (idx >> 6) & 15;
  int spos = (idx >> 10) & 1023;
  int bidx = idx >> 20;
  float inv = __expf(-(2.0f * (float)j / 128.0f) * 9.210340371976184f);  // ln(10000)
  float sn, cs;
  __sincosf((float)spos * inv, &sn, &cs);
  size_t inBase = (((size_t)(bidx * 1024 + spos)) * 16 + h) * 128;
  float x1 = q[inBase + j];
  float x2 = q[inBase + 64 + j];
  const float ss = 1.0f / 128.0f;
  size_t outBase = (((size_t)(bidx * 16 + h)) * 1024 + spos) * 128;
  qb[outBase + j]      = (__bf16)((x1 * cs - x2 * sn) * ss);
  qb[outBase + 64 + j] = (__bf16)((x1 * sn + x2 * cs) * ss);
}

// RoPE for K: [b,s,G,hd] fp32 -> [b,G,s,hd] bf16
__global__ void rope_k_kernel(const float* __restrict__ k, __bf16* __restrict__ kb) {
  int idx  = blockIdx.x * blockDim.x + threadIdx.x;  // < 2^20
  int j    = idx & 63;
  int g    = (idx >> 6) & 3;
  int spos = (idx >> 8) & 1023;
  int bidx = idx >> 18;
  float inv = __expf(-(2.0f * (float)j / 128.0f) * 9.210340371976184f);
  float sn, cs;
  __sincosf((float)spos * inv, &sn, &cs);
  size_t inBase = (((size_t)(bidx * 1024 + spos)) * 4 + g) * 128;
  float x1 = k[inBase + j];
  float x2 = k[inBase + 64 + j];
  size_t outBase = (((size_t)(bidx * 4 + g)) * 1024 + spos) * 128;
  kb[outBase + j]      = (__bf16)(x1 * cs - x2 * sn);
  kb[outBase + 64 + j] = (__bf16)(x1 * sn + x2 * cs);
}

// V relayout: [b,s,G,hd] fp32 -> [b,G,s,hd] bf16
__global__ void v_relayout_kernel(const float* __restrict__ v, __bf16* __restrict__ vb) {
  int idx  = blockIdx.x * blockDim.x + threadIdx.x;  // < 2^21
  int d    = idx & 127;
  int g    = (idx >> 7) & 3;
  int spos = (idx >> 9) & 1023;
  int bidx = idx >> 19;
  size_t inA  = (((size_t)(bidx * 1024 + spos)) * 4 + g) * 128 + d;
  size_t outA = (((size_t)(bidx * 4 + g)) * 1024 + spos) * 128 + d;
  vb[outA] = (__bf16)v[inA];
}

// ---------------------------------------------------------------------------
// Flash attention: grid (s/64, H, b), 128 threads = 4 waves, wave = 16 q-rows.
// Double-buffered K/V tiles; K staged via async global->LDS copies.
// ctx out layout [b, s, H, hd] fp32.
// ---------------------------------------------------------------------------
__global__ __launch_bounds__(128) void attn_kernel(
    const __bf16* __restrict__ qb, const __bf16* __restrict__ kb,
    const __bf16* __restrict__ vb, float* __restrict__ ctx) {
  __shared__ __align__(16) __bf16 Ks[2][32 * 136];    // [kk][d] padded
  __shared__ __align__(16) __bf16 Vt[2][128 * 40];    // [d][kk] padded (transposed)
  __shared__ __align__(16) __bf16 Ps[4 * 16 * 40];    // per-wave P tile 16x32

  const int tid  = threadIdx.x;
  const int lane = tid & 31;
  const int wave = tid >> 5;
  const int b = blockIdx.z;
  const int h = blockIdx.y;
  const int g = h >> 2;                      // R = H/G = 4
  const int qBase = blockIdx.x * 64 + wave * 16;

  const int lm    = lane & 15;
  const int half  = lane >> 4;
  const int koffA = half * 8;
  const int koffB = half * 8;

  // Staging coordinates: 128 threads cover 32 rows x 128 cols
  const int skk = tid >> 2;            // 0..31
  const int sdb = (tid & 3) * 32;      // 0,32,64,96

  const __bf16* kbase = kb + ((size_t)(b * 4 + g)) * 1024 * 128;
  const __bf16* vbase = vb + ((size_t)(b * 4 + g)) * 1024 * 128;
  const __bf16* kSrc = kbase + (size_t)skk * 128 + sdb;
  const __bf16* vSrc = vbase + (size_t)skk * 128 + sdb;

  // Preload Q A-fragments (16 rows x 128 = 4 chunks of K=32)
  AFrag aq[4];
  const __bf16* qrow = qb + (((size_t)(b * 16 + h)) * 1024 + qBase + lm) * 128;
#pragma unroll
  for (int c2 = 0; c2 < 4; ++c2) {
    aq[c2].h[0] = *(const v8bf*)(qrow + c2 * 32 + koffA);
    aq[c2].h[1] = *(const v8bf*)(qrow + c2 * 32 + koffA + 16);
  }

  v8f ctxAcc[8];
#pragma unroll
  for (int j = 0; j < 8; ++j) ctxAcc[j] = zero8();
  float rowMax[8], rowSum[8];
#pragma unroll
  for (int r = 0; r < 8; ++r) { rowMax[r] = -1e30f; rowSum[r] = 0.0f; }

  __bf16* ps = Ps + wave * 16 * 40;
  __bf16 vtmp[32];

  auto stageK = [&](int kv0, int buf) {
    const __bf16* src = kSrc + (size_t)kv0 * 128;
    __bf16* dst = &Ks[buf][skk * 136 + sdb];
#pragma unroll
    for (int q4 = 0; q4 < 4; ++q4) copy16B_g2l(src + q4 * 8, dst + q4 * 8);
  };
  auto loadV = [&](int kv0) {
    const uint4* s4 = (const uint4*)(vSrc + (size_t)kv0 * 128);
#pragma unroll
    for (int q4 = 0; q4 < 4; ++q4) *(uint4*)(vtmp + q4 * 8) = s4[q4];
  };
  auto storeV = [&](int buf) {
#pragma unroll
    for (int i = 0; i < 32; ++i) Vt[buf][(sdb + i) * 40 + skk] = vtmp[i];
  };

  // Prologue: fill buffer 0
  stageK(0, 0);
  loadV(0);
  storeV(0);
  async_wait_all();
  __syncthreads();

  int cur = 0;
  for (int kv0 = 0; kv0 < 1024; kv0 += 32) {
    const int nxt = cur ^ 1;
    const bool more = (kv0 + 32) < 1024;
    if (more) {
      stageK(kv0 + 32, nxt);   // async: overlaps with compute below
      loadV(kv0 + 32);
    }

    // Logits: two 16x16 tiles (key cols kv0..+15 and kv0+16..+31)
    v8f s0 = zero8(), s1 = zero8();
#pragma unroll
    for (int c2 = 0; c2 < 4; ++c2) {
      AFrag bk0, bk1;
      bk0.h[0] = *(const v8bf*)(&Ks[cur][lm * 136 + c2 * 32 + koffB]);
      bk0.h[1] = *(const v8bf*)(&Ks[cur][lm * 136 + c2 * 32 + koffB + 16]);
      bk1.h[0] = *(const v8bf*)(&Ks[cur][(16 + lm) * 136 + c2 * 32 + koffB]);
      bk1.h[1] = *(const v8bf*)(&Ks[cur][(16 + lm) * 136 + c2 * 32 + koffB + 16]);
      s0 = WMMA_BF16(aq[c2].v, bk0.v, s0);
      s1 = WMMA_BF16(aq[c2].v, bk1.v, s1);
    }

    // Online softmax (row reductions stay within the 16-lane half)
    float p0[8], p1[8], scl[8];
#pragma unroll
    for (int r = 0; r < 8; ++r) {
      float mx = fmaxf(s0[r], s1[r]);
#pragma unroll
      for (int m_ = 1; m_ < 16; m_ <<= 1) mx = fmaxf(mx, __shfl_xor(mx, m_, 32));
      float mNew = fmaxf(rowMax[r], mx);
      float sc = __expf(rowMax[r] - mNew);
      rowMax[r] = mNew;
      scl[r] = sc;
      float e0 = __expf(s0[r] - mNew);
      float e1 = __expf(s1[r] - mNew);
      p0[r] = e0; p1[r] = e1;
      float sum = e0 + e1;
#pragma unroll
      for (int m_ = 1; m_ < 16; m_ <<= 1) sum += __shfl_xor(sum, m_, 32);
      rowSum[r] = rowSum[r] * sc + sum;
    }
#pragma unroll
    for (int j = 0; j < 8; ++j)
#pragma unroll
      for (int r = 0; r < 8; ++r) ctxAcc[j][r] *= scl[r];

    // C-layout -> A-layout via per-wave LDS round trip
#pragma unroll
    for (int r = 0; r < 8; ++r) {
      int m = r + half * 8;
      ps[m * 40 + lm]      = (__bf16)p0[r];
      ps[m * 40 + 16 + lm] = (__bf16)p1[r];
    }
    AFrag pa;
    pa.h[0] = *(const v8bf*)(ps + lm * 40 + koffA);
    pa.h[1] = *(const v8bf*)(ps + lm * 40 + koffA + 16);

    // ctx += P (16x32) @ V (32x128)
#pragma unroll
    for (int j = 0; j < 8; ++j) {
      AFrag bv;
      bv.h[0] = *(const v8bf*)(&Vt[cur][(j * 16 + lm) * 40 + koffB]);
      bv.h[1] = *(const v8bf*)(&Vt[cur][(j * 16 + lm) * 40 + koffB + 16]);
      ctxAcc[j] = WMMA_BF16(pa.v, bv.v, ctxAcc[j]);
    }

    if (more) storeV(nxt);
    async_wait_all();
    __syncthreads();
    cur = nxt;
  }

  // Write normalized context: [b, s, H, hd]
#pragma unroll
  for (int j = 0; j < 8; ++j) {
#pragma unroll
    for (int r = 0; r < 8; ++r) {
      int m = r + half * 8;
      size_t addr = (((size_t)(b * 1024 + qBase + m)) * 16 + h) * 128 + j * 16 + lm;
      ctx[addr] = ctxAcc[j][r] / rowSum[r];
    }
  }
}

// ---------------------------------------------------------------------------
// Host launch
// ---------------------------------------------------------------------------
extern "C" void kernel_launch(void* const* d_in, const int* in_sizes, int n_in,
                              void* d_out, int out_size, void* d_ws, size_t ws_size,
                              hipStream_t stream) {
  const float* x  = (const float*)d_in[0];
  const float* Wq = (const float*)d_in[1];
  const float* bq = (const float*)d_in[2];
  const float* Wk = (const float*)d_in[3];
  const float* bk = (const float*)d_in[4];
  const float* Wv = (const float*)d_in[5];
  const float* bv = (const float*)d_in[6];
  const float* Wo = (const float*)d_in[7];
  const float* bo = (const float*)d_in[8];
  float* out = (float*)d_out;

  const int BT = 4 * 1024;  // tokens
  char* ws = (char*)d_ws;
  size_t off = 0;
  auto alloc = [&](size_t bytes) -> void* {
    void* p = ws + off;
    off += (bytes + 255) & ~(size_t)255;
    return p;
  };
  __bf16* xb  = (__bf16*)alloc((size_t)BT * 2048 * 2);
  __bf16* Wqb = (__bf16*)alloc((size_t)2048 * 2048 * 2);
  __bf16* Wkb = (__bf16*)alloc((size_t)2048 * 512 * 2);
  __bf16* Wvb = (__bf16*)alloc((size_t)2048 * 512 * 2);
  __bf16* Wob = (__bf16*)alloc((size_t)2048 * 2048 * 2);
  float*  qf  = (float*)alloc((size_t)BT * 2048 * 4);
  float*  kf  = (float*)alloc((size_t)BT * 512 * 4);
  float*  vf  = (float*)alloc((size_t)BT * 512 * 4);
  __bf16* qbb = (__bf16*)alloc((size_t)BT * 2048 * 2);
  __bf16* kbb = (__bf16*)alloc((size_t)BT * 512 * 2);
  __bf16* vbb = (__bf16*)alloc((size_t)BT * 512 * 2);
  float*  ctx  = qf;   // reuse: qf dead after rope_q
  __bf16* ctxb = xb;   // reuse: xb dead after projections

  auto cdiv = [](int a, int b) { return (a + b - 1) / b; };
  int n;
  n = BT * 2048;   cast_f32_bf16<<<cdiv(n, 256), 256, 0, stream>>>(x,  xb,  n);
  n = 2048 * 2048; cast_f32_bf16<<<cdiv(n, 256), 256, 0, stream>>>(Wq, Wqb, n);
  n = 2048 * 512;  cast_f32_bf16<<<cdiv(n, 256), 256, 0, stream>>>(Wk, Wkb, n);
  n = 2048 * 512;  cast_f32_bf16<<<cdiv(n, 256), 256, 0, stream>>>(Wv, Wvb, n);
  n = 2048 * 2048; cast_f32_bf16<<<cdiv(n, 256), 256, 0, stream>>>(Wo, Wob, n);

  dim3 gq(2048 / 128, BT / 128);
  dim3 gkv(512 / 128, BT / 128);
  gemm_bf16_wmma<<<gq,  256, 0, stream>>>(xb, Wqb, bq, qf, BT, 2048, 2048);
  gemm_bf16_wmma<<<gkv, 256, 0, stream>>>(xb, Wkb, bk, kf, BT, 512, 2048);
  gemm_bf16_wmma<<<gkv, 256, 0, stream>>>(xb, Wvb, bv, vf, BT, 512, 2048);

  rope_q_kernel<<<(4 * 16 * 1024 * 64) / 256, 256, 0, stream>>>(qf, qbb);
  rope_k_kernel<<<(4 * 4 * 1024 * 64) / 256, 256, 0, stream>>>(kf, kbb);
  v_relayout_kernel<<<(4 * 4 * 1024 * 128) / 256, 256, 0, stream>>>(vf, vbb);

  dim3 ga(1024 / 64, 16, 4);
  attn_kernel<<<ga, 128, 0, stream>>>(qbb, kbb, vbb, ctx);

  n = BT * 2048;
  cast_f32_bf16<<<cdiv(n, 256), 256, 0, stream>>>(ctx, ctxb, n);
  gemm_bf16_wmma<<<gq, 256, 0, stream>>>(ctxb, Wob, bo, out, BT, 2048, 2048);
}
